// LSTMCharacter_44822278701288
// MI455X (gfx1250) — matrix operand order repaired and checked by reading, not compiled, not run on hardware
//
#include <hip/hip_runtime.h>
#include <hip/hip_bf16.h>

// ---------------------------------------------------------------------------
// Bidirectional char-LSTM, fused per-step GEMM (bf16 WMMA, f32 accum) + gates.
// B=4096, T=32, H=512, E=64, O=64.
// ---------------------------------------------------------------------------

#define BB 4096
#define TT 32
#define HH 512
#define EE 64
#define OO 64

typedef __attribute__((ext_vector_type(16))) __bf16 v16bf;
typedef __attribute__((ext_vector_type(8)))  float  v8f;

union Frag {
    uint4  u[2];
    v16bf  bf;
};

__device__ __forceinline__ unsigned short f2bf(float f) {
    unsigned int u = __float_as_uint(f);
    return (unsigned short)((u + 0x7FFFu + ((u >> 16) & 1u)) >> 16);
}
__device__ __forceinline__ float bf2f(unsigned short h) {
    return __uint_as_float(((unsigned int)h) << 16);
}
__device__ __forceinline__ float sigmoidf(float x) {
    return 1.0f / (1.0f + __expf(-x));
}
__device__ __forceinline__ float tanh_fast(float x) {
    // 1 - 2/(e^{2x}+1): monotone, saturates cleanly to +/-1, no inf/inf NaN.
    return 1.0f - 2.0f / (__expf(2.0f * x) + 1.0f);
}

// ---------------------------------------------------------------------------
// Preprocessing kernels
// ---------------------------------------------------------------------------
__global__ void f32_to_bf16_kernel(const float* __restrict__ in,
                                   unsigned short* __restrict__ out, int n) {
    int i = blockIdx.x * blockDim.x + threadIdx.x;
    if (i < n) out[i] = f2bf(in[i]);
}

__global__ void bias_add_kernel(const float* __restrict__ a,
                                const float* __restrict__ b,
                                float* __restrict__ o, int n) {
    int i = blockIdx.x * blockDim.x + threadIdx.x;
    if (i < n) o[i] = a[i] + b[i];
}

__global__ void zero_u32_kernel(unsigned int* __restrict__ p, long n) {
    long i = (long)blockIdx.x * blockDim.x + threadIdx.x;
    long stride = (long)gridDim.x * blockDim.x;
    for (; i < n; i += stride) p[i] = 0u;
}

// ---------------------------------------------------------------------------
// Fused LSTM step: gates = emb[x[:,t]] @ Wih^T + h_in @ Whh^T + bias,
// then c/h update. One block = 64 batch rows x 64 hidden cols (x 4 gates).
// 8 waves: wave = rstrip(4) x colpair(2). Each wave: 4 gates x 2 colstrips
// of 16x16 WMMA tiles -> i/f/g/o for the same (b,j) live in the same
// lane/VGPR slot across the 4 gate accumulators.
// All weight-row base pointers are hoisted out of the K loops and the K loops
// are fully unrolled so in-loop addresses are (base + constant) -> immediate
// offsets, no VALU address math between WMMAs (avoids WMMA->VALU hazard NOPs).
// gridDim.z selects cell (0: forward time, 1: reversed time).
// ---------------------------------------------------------------------------
__global__ __launch_bounds__(256) void lstm_step_kernel(
    const int* __restrict__ x,
    const unsigned short* __restrict__ emb,
    const unsigned short* __restrict__ Wih1, const unsigned short* __restrict__ Whh1,
    const float* __restrict__ bias1,
    const unsigned short* __restrict__ Wih2, const unsigned short* __restrict__ Whh2,
    const float* __restrict__ bias2,
    const unsigned short* __restrict__ h1_in, unsigned short* __restrict__ h1_out,
    float* __restrict__ c1,
    const unsigned short* __restrict__ h2_in, unsigned short* __restrict__ h2_out,
    float* __restrict__ c2,
    int t)
{
    const int cell = blockIdx.z;
    const unsigned short* Wih  = cell ? Wih2 : Wih1;
    const unsigned short* Whh  = cell ? Whh2 : Whh1;
    const float*          bias = cell ? bias2 : bias1;
    const unsigned short* h_in = cell ? h2_in : h1_in;
    unsigned short*      h_out = cell ? h2_out : h1_out;
    float*                   c = cell ? c2 : c1;
    const int t_eff = cell ? (TT - 1 - t) : t;

    const int lane   = threadIdx.x & 31;
    const int wave   = threadIdx.x >> 5;
    const int rstrip = wave & 3;    // which 16-row strip of the 64-row block tile
    const int cpair  = wave >> 2;   // which pair of 16-col strips
    const int laneLo = lane & 15;
    const int laneHi = lane >> 4;

    const int row0 = blockIdx.x * 64 + rstrip * 16;  // batch-row base of tile
    const int arow = row0 + laneLo;                  // row this lane feeds to A
    const int jblk = blockIdx.y * 64;                // hidden-col base of block
    const int col0 = jblk + 2 * cpair * 16 + laneLo; // weight row (cc=0)

    v8f acc[4][2];
#pragma unroll
    for (int g = 0; g < 4; ++g)
#pragma unroll
        for (int cc = 0; cc < 2; ++cc)
            acc[g][cc] = (v8f){0.f, 0.f, 0.f, 0.f, 0.f, 0.f, 0.f, 0.f};

    // ---- Phase 1: x-part, K = 64 (embedding row gathered per batch row) ----
    {
        const int idx = x[arow * TT + t_eff];
        // A base, pre-biased by this lane's K sub-offset (A 16x32 bf16 layout)
        const unsigned short* aptr = emb + (size_t)idx * EE + laneHi * 8;
        // B bases, pre-biased by this lane's K sub-offset (B 32x16 bf16 layout)
        const unsigned short* bp[8];
#pragma unroll
        for (int g = 0; g < 4; ++g)
#pragma unroll
            for (int cc = 0; cc < 2; ++cc)
                bp[g * 2 + cc] =
                    Wih + (size_t)(g * HH + col0 + cc * 16) * EE + laneHi * 8;

#pragma unroll
        for (int kc = 0; kc < EE; kc += 32) {
            Frag a;
            a.u[0] = *(const uint4*)(aptr + kc);
            a.u[1] = *(const uint4*)(aptr + kc + 16);
#pragma unroll
            for (int g = 0; g < 4; ++g) {
#pragma unroll
                for (int cc = 0; cc < 2; ++cc) {
                    Frag b;
                    b.u[0] = *(const uint4*)(bp[g * 2 + cc] + kc);
                    b.u[1] = *(const uint4*)(bp[g * 2 + cc] + kc + 16);
                    acc[g][cc] = __builtin_amdgcn_wmma_f32_16x16x32_bf16(
                        false, a.bf, false, b.bf, (short)0, acc[g][cc], false, false);
                }
            }
        }
    }

    // ---- Phase 2: h-part, K = 512, fully unrolled ----
    {
        const unsigned short* aptr = h_in + (size_t)arow * HH + laneHi * 8;
        const unsigned short* bp[8];
#pragma unroll
        for (int g = 0; g < 4; ++g)
#pragma unroll
            for (int cc = 0; cc < 2; ++cc)
                bp[g * 2 + cc] =
                    Whh + (size_t)(g * HH + col0 + cc * 16) * HH + laneHi * 8;

#pragma unroll
        for (int kc = 0; kc < HH; kc += 32) {
            Frag a;
            a.u[0] = *(const uint4*)(aptr + kc);
            a.u[1] = *(const uint4*)(aptr + kc + 16);
#pragma unroll
            for (int g = 0; g < 4; ++g) {
#pragma unroll
                for (int cc = 0; cc < 2; ++cc) {
                    Frag b;
                    b.u[0] = *(const uint4*)(bp[g * 2 + cc] + kc);
                    b.u[1] = *(const uint4*)(bp[g * 2 + cc] + kc + 16);
                    acc[g][cc] = __builtin_amdgcn_wmma_f32_16x16x32_bf16(
                        false, a.bf, false, b.bf, (short)0, acc[g][cc], false, false);
                }
            }
        }
    }

    // ---- Epilogue: C/D layout -> element (m = v + 8*laneHi, n = laneLo) ----
#pragma unroll
    for (int cc = 0; cc < 2; ++cc) {
        const int j = col0 + cc * 16;
        const float bi = bias[j];
        const float bf_ = bias[HH + j];
        const float bg = bias[2 * HH + j];
        const float bo = bias[3 * HH + j];
#pragma unroll
        for (int v = 0; v < 8; ++v) {
            const int m = row0 + laneHi * 8 + v;
            const size_t o = (size_t)m * HH + j;
            const float iv = sigmoidf(acc[0][cc][v] + bi);
            const float fv = sigmoidf(acc[1][cc][v] + bf_);
            const float gv = tanh_fast(acc[2][cc][v] + bg);
            const float ov = sigmoidf(acc[3][cc][v] + bo);
            const float cn = fv * c[o] + iv * gv;
            c[o] = cn;
            h_out[o] = f2bf(ov * tanh_fast(cn));
        }
    }
}

// ---------------------------------------------------------------------------
// Final: out = softmax([h1|h2] @ Wlin^T + blin). One block (64 threads) / row.
// ---------------------------------------------------------------------------
__global__ __launch_bounds__(64) void final_linear_softmax_kernel(
    const unsigned short* __restrict__ h1, const unsigned short* __restrict__ h2,
    const float* __restrict__ Wlin, const float* __restrict__ blin,
    float* __restrict__ out)
{
    __shared__ float hrow[2 * HH];
    __shared__ float logits[OO];
    const int b = blockIdx.x;
    const int tid = threadIdx.x;

    for (int k = tid; k < HH; k += OO) {
        hrow[k]      = bf2f(h1[(size_t)b * HH + k]);
        hrow[HH + k] = bf2f(h2[(size_t)b * HH + k]);
    }
    __syncthreads();

    const float* w = Wlin + (size_t)tid * (2 * HH);
    float s = blin[tid];
#pragma unroll 4
    for (int k = 0; k < 2 * HH; k += 4) {
        s += hrow[k] * w[k] + hrow[k + 1] * w[k + 1] +
             hrow[k + 2] * w[k + 2] + hrow[k + 3] * w[k + 3];
    }
    logits[tid] = s;
    __syncthreads();

    float mx = -1e30f;
    for (int i = 0; i < OO; ++i) mx = fmaxf(mx, logits[i]);
    float sum = 0.f;
    for (int i = 0; i < OO; ++i) sum += __expf(logits[i] - mx);
    out[(size_t)b * OO + tid] = __expf(s - mx) / sum;
}

// ---------------------------------------------------------------------------
// Host side
// ---------------------------------------------------------------------------
extern "C" void kernel_launch(void* const* d_in, const int* in_sizes, int n_in,
                              void* d_out, int out_size, void* d_ws, size_t ws_size,
                              hipStream_t stream) {
    const int*   x     = (const int*)d_in[0];
    // d_in[1] = mask (unused by reference math)
    const float* emb   = (const float*)d_in[2];
    const float* Wih1  = (const float*)d_in[3];
    const float* Whh1  = (const float*)d_in[4];
    const float* bih1  = (const float*)d_in[5];
    const float* bhh1  = (const float*)d_in[6];
    const float* Wih2  = (const float*)d_in[7];
    const float* Whh2  = (const float*)d_in[8];
    const float* bih2  = (const float*)d_in[9];
    const float* bhh2  = (const float*)d_in[10];
    const float* Wlin  = (const float*)d_in[11];
    const float* blin  = (const float*)d_in[12];
    float* out = (float*)d_out;

    char* ws = (char*)d_ws;
    size_t off = 0;
    auto alloc = [&](size_t bytes) -> void* {
        void* p = ws + off;
        off = (off + bytes + 255) & ~(size_t)255;
        return p;
    };

    unsigned short* emb_bf  = (unsigned short*)alloc((size_t)256 * EE * 2);
    unsigned short* Wih1_bf = (unsigned short*)alloc((size_t)4 * HH * EE * 2);
    unsigned short* Wih2_bf = (unsigned short*)alloc((size_t)4 * HH * EE * 2);
    unsigned short* Whh1_bf = (unsigned short*)alloc((size_t)4 * HH * HH * 2);
    unsigned short* Whh2_bf = (unsigned short*)alloc((size_t)4 * HH * HH * 2);
    float*          b1      = (float*)alloc((size_t)4 * HH * 4);
    float*          b2      = (float*)alloc((size_t)4 * HH * 4);
    float*          c1      = (float*)alloc((size_t)BB * HH * 4);
    float*          c2      = (float*)alloc((size_t)BB * HH * 4);
    unsigned short* h1a     = (unsigned short*)alloc((size_t)BB * HH * 2);
    unsigned short* h1b     = (unsigned short*)alloc((size_t)BB * HH * 2);
    unsigned short* h2a     = (unsigned short*)alloc((size_t)BB * HH * 2);
    unsigned short* h2b     = (unsigned short*)alloc((size_t)BB * HH * 2);

    // --- preprocess: bf16 conversions + fused biases + state zeroing ---
    {
        int n;
        n = 256 * EE;
        f32_to_bf16_kernel<<<(n + 255) / 256, 256, 0, stream>>>(emb, emb_bf, n);
        n = 4 * HH * EE;
        f32_to_bf16_kernel<<<(n + 255) / 256, 256, 0, stream>>>(Wih1, Wih1_bf, n);
        f32_to_bf16_kernel<<<(n + 255) / 256, 256, 0, stream>>>(Wih2, Wih2_bf, n);
        n = 4 * HH * HH;
        f32_to_bf16_kernel<<<(n + 255) / 256, 256, 0, stream>>>(Whh1, Whh1_bf, n);
        f32_to_bf16_kernel<<<(n + 255) / 256, 256, 0, stream>>>(Whh2, Whh2_bf, n);
        n = 4 * HH;
        bias_add_kernel<<<(n + 255) / 256, 256, 0, stream>>>(bih1, bhh1, b1, n);
        bias_add_kernel<<<(n + 255) / 256, 256, 0, stream>>>(bih2, bhh2, b2, n);

        const long cw = (long)BB * HH;      // f32 words
        const long hw = (long)BB * HH / 2;  // bf16 halves -> u32 words
        zero_u32_kernel<<<2048, 256, 0, stream>>>((unsigned int*)c1, cw);
        zero_u32_kernel<<<2048, 256, 0, stream>>>((unsigned int*)c2, cw);
        zero_u32_kernel<<<2048, 256, 0, stream>>>((unsigned int*)h1a, hw);
        zero_u32_kernel<<<2048, 256, 0, stream>>>((unsigned int*)h2a, hw);
    }

    // --- recurrent steps: ping-pong h buffers (t=31 writes the 'a' buffers) ---
    dim3 sgrid(BB / 64, HH / 64, 2);
    for (int t = 0; t < TT; ++t) {
        const unsigned short* h1_in  = (t & 1) ? h1b : h1a;
        unsigned short*       h1_out = (t & 1) ? h1a : h1b;
        const unsigned short* h2_in  = (t & 1) ? h2b : h2a;
        unsigned short*       h2_out = (t & 1) ? h2a : h2b;
        lstm_step_kernel<<<sgrid, 256, 0, stream>>>(
            x, emb_bf,
            Wih1_bf, Whh1_bf, b1,
            Wih2_bf, Whh2_bf, b2,
            h1_in, h1_out, c1,
            h2_in, h2_out, c2,
            t);
    }

    // --- output head ---
    final_linear_softmax_kernel<<<BB, OO, 0, stream>>>(h1a, h2a, Wlin, blin, out);
}
